// NonLocalSpatialPropagation_25821343384347
// MI455X (gfx1250) — matrix operand order repaired and checked by reading, not compiled, not run on hardware
//
#include <hip/hip_runtime.h>
#include <hip/hip_bf16.h>
#include <math.h>
#include <stdint.h>

typedef __attribute__((ext_vector_type(16))) _Float16 v16h;
typedef __attribute__((ext_vector_type(8)))  _Float16 v8h;
typedef __attribute__((ext_vector_type(8)))  float    v8f;

#define B_   4
#define H_   384
#define W_   1280
#define NPIX (B_*H_*W_)          // 1,966,080
#define TPR  (W_/16)             // 80 pixel-tiles per row (exact)
#define NTILES (B_*H_*TPR)       // 122,880 (divisible by 8)
#define SEGS (TPR/8)             // 10 x 128-pixel segments per row (exact)

// gfx1250 async global->LDS copy path (ASYNCcnt), with safe fallback.
// Probe result (round 2 diagnostic): param 0 is
//   'int __attribute__((vector_size(16))) __device__ *'  (AS1, int4, non-const)
#if defined(__gfx1250__) && __has_builtin(__builtin_amdgcn_global_load_async_to_lds_b128)
  #define ASYNC_LDS 1
  typedef int async_v4i __attribute__((vector_size(16)));
  typedef __attribute__((address_space(1))) async_v4i as1_v4i;
  typedef __attribute__((address_space(3))) async_v4i as3_v4i;
#else
  #define ASYNC_LDS 0
#endif

__device__ inline v8h v8h_zero() {
  v8h z;
  #pragma unroll
  for (int i = 0; i < 8; ++i) z[i] = (_Float16)0.f;
  return z;
}

__device__ inline void copy16_to_lds(_Float16* dst, const _Float16* src) {
#if ASYNC_LDS
  __builtin_amdgcn_global_load_async_to_lds_b128(
      (as1_v4i*)(uintptr_t)src, (as3_v4i*)(uintptr_t)dst, 0, 0);
#else
  *(v8h*)dst = *(const v8h*)src;
#endif
}

__device__ inline void async_copy_fence() {
#if ASYNC_LDS
  #if __has_builtin(__builtin_amdgcn_s_wait_asynccnt)
    __builtin_amdgcn_s_wait_asynccnt(0);
  #else
    asm volatile("s_wait_asynccnt 0x0" ::: "memory");
  #endif
#endif
}

// ---------------------------------------------------------------------------
// Pack OIHW f32 weights into this code's WMMA A-operand layout (contiguous k order), f16.
// A (16x32 f16): lane l -> row m = l%16; half e: k=(l<16?0:16)+e.
//                K-global = step*32 + k.
// ---------------------------------------------------------------------------
__global__ void pack_weights_kernel(const float* __restrict__ W,
                                    _Float16* __restrict__ out,
                                    int tiles, int steps, int grp,
                                    int cinReal, int outReal) {
  int idx = blockIdx.x * 256 + threadIdx.x;
  int total = tiles * steps * 32 * 16;
  if (idx >= total) return;
  int e = idx & 15;
  int l = (idx >> 4) & 31;
  int rest = idx >> 9;
  int s = rest % steps;
  int t = rest / steps;
  int outc = t * 16 + (l & 15);
  int khalf = (l < 16 ? 0 : 16) + e;
  int kg = s * 32 + khalf;
  int tap = kg / grp;
  int ch  = kg % grp;
  float v = 0.f;
  if (tap < 9 && ch < cinReal && outc < outReal)
    v = W[(outc * cinReal + ch) * 9 + tap];     // OIHW, tap = ky*3+kx
  out[idx] = (_Float16)v;
}

__global__ void bn_prep_kernel(const float* __restrict__ g, const float* __restrict__ b,
                               const float* __restrict__ m, const float* __restrict__ v,
                               float* __restrict__ scale, float* __restrict__ bias, int C) {
  int c = blockIdx.x * 64 + threadIdx.x;
  if (c >= C) return;
  float s = g[c] * rsqrtf(v[c] + 1e-5f);
  scale[c] = s;
  bias[c]  = b[c] - m[c] * s;
}

// ---------------------------------------------------------------------------
// Stage 1: disparity warp + guidance pack (NHWC f16, 12 real + 4 zero pad ch)
// ---------------------------------------------------------------------------
__global__ void guidance_kernel(const float* __restrict__ disp,
                                const float* __restrict__ normal,
                                const float* __restrict__ left,
                                const float* __restrict__ right,
                                _Float16* __restrict__ gout) {
  int pix = blockIdx.x * 256 + threadIdx.x;
  if (pix >= NPIX) return;
  int x = pix % W_;
  int y = (pix / W_) % H_;
  int b = pix / (W_ * H_);
  const int plane = H_ * W_;
  int rowbase = y * W_;

  float d  = disp[pix];
  float xs = (float)x - d;
  float x0 = floorf(xs);
  int   x0i = (int)x0;
  float w1 = xs - x0;
  float w0 = 1.f - w1;

  union { v8h v[2]; _Float16 h[16]; } u;
  #pragma unroll
  for (int c = 0; c < 3; ++c) {
    const float* rp = right + ((size_t)(b * 3 + c)) * plane + rowbase;
    float acc = 0.f;
    int xi0 = x0i, xi1 = x0i + 1;
    if (xi0 >= 0 && xi0 < W_) acc += w0 * rp[xi0];
    if (xi1 >= 0 && xi1 < W_) acc += w1 * rp[xi1];
    float nl = normal[((size_t)(b * 3 + c)) * plane + rowbase + x];
    float lf = left  [((size_t)(b * 3 + c)) * plane + rowbase + x];
    float rg = right [((size_t)(b * 3 + c)) * plane + rowbase + x];
    u.h[c]     = (_Float16)nl;
    u.h[3 + c] = (_Float16)lf;
    u.h[6 + c] = (_Float16)rg;
    u.h[9 + c] = (_Float16)(acc - lf);   // warped_error
  }
  u.h[12] = u.h[13] = u.h[14] = u.h[15] = (_Float16)0.f;

  v8h* dst = (v8h*)(gout + (size_t)pix * 16);
  dst[0] = u.v[0];
  dst[1] = u.v[1];
}

// ---------------------------------------------------------------------------
// Implicit-GEMM 3x3 conv via v_wmma_f32_16x16x32_f16, LDS-staged activations.
//   Block = 8 waves = 128 contiguous pixels of one image row.
//   LDS holds: packed weights (A-layout) + 3-row x 130-pixel activation halo.
//   Wave tile: D(16 outC x 16 pixels). B from LDS (ds_load_b128 pairs).
//   MODE 1: input 16 halfs/pix, K-step = two 16-ch taps  (conv1: 12->32)
//   MODE 2: input 32 halfs/pix, K-step = one 32-ch tap   (conv2: 32->64)
//   MODE 3: input 64 halfs/pix, K-step = half a 64-ch tap(conv3: 64->24pad32)
// ---------------------------------------------------------------------------
#define ACT_W 132   // staged pixels per row (0..129 used; halo = x0-1..x0+128)

template<int MODE, int TILES, int STEPS, int CIN_PAD, int COUT_PAD, bool BN>
__global__ void conv_wmma_kernel(const _Float16* __restrict__ in,
                                 const _Float16* __restrict__ wpacked,
                                 const float* __restrict__ bnscale,
                                 const float* __restrict__ bnbias,
                                 _Float16* __restrict__ out) {
  __shared__ __align__(16) _Float16 wlds[TILES * STEPS * 32 * 16];
  __shared__ __align__(16) _Float16 act[3 * ACT_W * CIN_PAD];

  // block -> (image, row, 128-pixel segment)
  const int seg = blockIdx.x % SEGS;
  const int rowIdx = blockIdx.x / SEGS;
  const int y  = rowIdx % H_;
  const int bb = rowIdx / H_;
  const int x0 = seg * 128;

  // ---- stage packed weights: global -> LDS (async path) ----
  const int nvec = (TILES * STEPS * 32 * 16) / 8;
  for (int i = threadIdx.x; i < nvec; i += 256)
    copy16_to_lds(wlds + i * 8, wpacked + i * 8);

  // ---- stage 3-row activation halo: global -> LDS, zero OOB chunks ----
  const int CH8 = CIN_PAD / 8;            // 16B chunks per pixel
  const int chunksPerRow = 130 * CH8;
  for (int i = threadIdx.x; i < 3 * chunksPerRow; i += 256) {
    int r  = i / chunksPerRow;
    int j  = i - r * chunksPerRow;
    int p  = j / CH8;                      // 0..129  (pixel x0-1+p)
    int c8 = j - p * CH8;
    int yy = y + r - 1;
    int xx = x0 - 1 + p;
    _Float16* dst = act + ((r * ACT_W + p) * CIN_PAD) + c8 * 8;
    if (yy >= 0 && yy < H_ && xx >= 0 && xx < W_) {
      const _Float16* src =
          in + ((size_t)((bb * H_ + yy) * W_ + xx)) * CIN_PAD + c8 * 8;
      copy16_to_lds(dst, src);
    } else {
      *(v8h*)dst = v8h_zero();
    }
  }
  async_copy_fence();
  __syncthreads();

  const int lane = threadIdx.x & 31;
  const int wave = threadIdx.x >> 5;
  const int hi   = lane >> 4;              // 0: K 0..15, 1: K 16..31 (B halves)
  const int pl   = (wave << 4) + (lane & 15) + 1;   // lane pixel in halo coords

  v8f acc[TILES];
  #pragma unroll
  for (int t = 0; t < TILES; ++t)
    #pragma unroll
    for (int r = 0; r < 8; ++r) acc[t][r] = 0.f;

  #pragma unroll
  for (int s = 0; s < STEPS; ++s) {
    // ---- B operand from LDS halo ----
    int tap, chbase;
    if (MODE == 1)      { tap = 2 * s + hi;  chbase = 0; }
    else if (MODE == 2) { tap = s;           chbase = hi * 16; }
    else                { tap = s >> 1;      chbase = (s & 1) * 32 + hi * 16; }

    union { v16h v; v8h h[2]; } bmat;
    if (MODE == 1 && tap >= 9) {             // conv1's zero-padded 10th tap
      bmat.h[0] = v8h_zero();
      bmat.h[1] = v8h_zero();
    } else {
      int r   = tap / 3;                     // halo row 0..2
      int pnb = pl + (tap % 3) - 1;          // neighbor pixel, 0..129
      const _Float16* bsrc = act + ((r * ACT_W + pnb) * CIN_PAD) + chbase;
      bmat.h[0] = *(const v8h*)bsrc;
      bmat.h[1] = *(const v8h*)(bsrc + 8);
    }

    // ---- A operands from LDS + WMMA per output-channel tile ----
    #pragma unroll
    for (int t = 0; t < TILES; ++t) {
      const v16h* ap =
          (const v16h*)(wlds + (((t * STEPS + s) * 32 + lane) << 4));
      acc[t] = __builtin_amdgcn_wmma_f32_16x16x32_f16(
          false, *ap, false, bmat.v, (short)0, acc[t], false, false);
    }
  }

  // ---- epilogue: (BN + ReLU) -> f16 NHWC store ----
  int x = x0 + (wave << 4) + (lane & 15);
  size_t pix = (size_t)(bb * H_ + y) * W_ + x;
  #pragma unroll
  for (int t = 0; t < TILES; ++t) {
    int cbase = t * 16 + hi * 8;             // D: VGPR r -> M = r + hi*8
    union { v8h v; _Float16 h[8]; } o;
    #pragma unroll
    for (int r = 0; r < 8; ++r) {
      float val = acc[t][r];
      if constexpr (BN) {
        val = fmaxf(val * bnscale[cbase + r] + bnbias[cbase + r], 0.f);
      }
      o.h[r] = (_Float16)val;
    }
    *(v8h*)(out + pix * COUT_PAD + cbase) = o.v;
  }
}

// ---------------------------------------------------------------------------
// Stage 3: non-local propagation (gather-heavy, scalar f32)
// ---------------------------------------------------------------------------
__device__ inline float bilin_plane(const float* __restrict__ p, float yf, float xf) {
  float y0 = floorf(yf), x0 = floorf(xf);
  int y0i = (int)y0, x0i = (int)x0;
  float wy1 = yf - y0, wx1 = xf - x0;
  float acc = 0.f;
  #pragma unroll
  for (int dy = 0; dy < 2; ++dy) {
    int yy = y0i + dy;
    if (yy < 0 || yy >= H_) continue;
    float wy = dy ? wy1 : 1.f - wy1;
    #pragma unroll
    for (int dx = 0; dx < 2; ++dx) {
      int xx = x0i + dx;
      if (xx < 0 || xx >= W_) continue;
      float wx = dx ? wx1 : 1.f - wx1;
      acc += wy * wx * p[yy * W_ + xx];
    }
  }
  return acc;
}

__global__ void final_kernel(const _Float16* __restrict__ y3,
                             const float* __restrict__ disp,
                             const float* __restrict__ conf,
                             const float* __restrict__ affScale,
                             float* __restrict__ outp) {
  int pix = blockIdx.x * 256 + threadIdx.x;
  if (pix >= NPIX) return;
  int x = pix % W_;
  int y = (pix / W_) % H_;
  int b = pix / (H_ * W_);

  const _Float16* src = y3 + (size_t)pix * 32;
  float o1[8], o2[8], ar[8];
  #pragma unroll
  for (int j = 0; j < 8; ++j) {
    o1[j] = (float)src[j];          // y-offsets
    o2[j] = (float)src[8 + j];      // x-offsets
    ar[j] = (float)src[16 + j];     // raw affinity
  }

  const float* dPlane = disp + (size_t)b * H_ * W_;
  const float* cPlane = conf + (size_t)b * H_ * W_;
  float invA = 1.f / (affScale[0] + 1e-8f);

  float aff[8], ssum = 0.f;
  #pragma unroll
  for (int j = 0; j < 8; ++j) {
    float ca = bilin_plane(cPlane, (float)y + o1[j], (float)x + o2[j]);
    float a  = tanhf(ar[j]) * invA * ca;
    aff[j] = a;
    ssum  += fabsf(a);
  }
  float s = fmaxf(ssum + 1e-4f, 1.f);
  float asum = 0.f;
  #pragma unroll
  for (int j = 0; j < 8; ++j) { aff[j] /= s; asum += aff[j]; }
  float affRef = 1.f - asum;

  float inter = 0.f;
  #pragma unroll
  for (int c = 0; c < 9; ++c) {
    float w, dy, dx;
    if (c == 4) { w = affRef; dy = 0.f; dx = 0.f; }
    else { int j = (c < 4) ? c : c - 1; w = aff[j]; dy = o1[j]; dx = o2[j]; }
    float ky = (float)(c / 3) - 1.f;
    float kx = (float)(c % 3) - 1.f;
    inter += w * bilin_plane(dPlane, (float)y + ky + dy, (float)x + kx + dx);
  }
  inter = fmaxf(inter, 0.f);
  outp[pix] = fmaxf(0.7f * disp[pix] + 0.3f * inter, 0.f);
}

// ---------------------------------------------------------------------------
extern "C" void kernel_launch(void* const* d_in, const int* in_sizes, int n_in,
                              void* d_out, int out_size, void* d_ws, size_t ws_size,
                              hipStream_t stream) {
  (void)in_sizes; (void)n_in; (void)out_size; (void)ws_size;

  const float* disp   = (const float*)d_in[0];
  const float* normal = (const float*)d_in[1];
  const float* left   = (const float*)d_in[2];
  const float* right  = (const float*)d_in[3];
  const float* conf   = (const float*)d_in[4];
  const float* w1     = (const float*)d_in[5];
  const float* g1 = (const float*)d_in[6];  const float* b1 = (const float*)d_in[7];
  const float* m1 = (const float*)d_in[8];  const float* v1 = (const float*)d_in[9];
  const float* w2     = (const float*)d_in[10];
  const float* g2 = (const float*)d_in[11]; const float* b2 = (const float*)d_in[12];
  const float* m2 = (const float*)d_in[13]; const float* v2 = (const float*)d_in[14];
  const float* w3     = (const float*)d_in[15];
  const float* affS   = (const float*)d_in[16];
  float* outp = (float*)d_out;

  char* ws = (char*)d_ws;
  size_t off = 0;
  auto alloc = [&](size_t bytes) -> char* {
    char* p = ws + off;
    off = (off + bytes + 255) & ~(size_t)255;
    return p;
  };
  _Float16* guid = (_Float16*)alloc((size_t)NPIX * 16 * 2);   // 63 MB
  _Float16* x1   = (_Float16*)alloc((size_t)NPIX * 32 * 2);   // 126 MB
  _Float16* x2   = (_Float16*)alloc((size_t)NPIX * 64 * 2);   // 252 MB
  _Float16* y3   = (_Float16*)alloc((size_t)NPIX * 32 * 2);   // 126 MB
  _Float16* pw1  = (_Float16*)alloc((size_t)2 * 5  * 512 * 2);
  _Float16* pw2  = (_Float16*)alloc((size_t)4 * 9  * 512 * 2);
  _Float16* pw3  = (_Float16*)alloc((size_t)2 * 18 * 512 * 2);
  float* bns1 = (float*)alloc(32 * 4); float* bnb1 = (float*)alloc(32 * 4);
  float* bns2 = (float*)alloc(64 * 4); float* bnb2 = (float*)alloc(64 * 4);

  // weight repack + BN fold (tiny)
  pack_weights_kernel<<<(2 * 5  * 512 + 255) / 256, 256, 0, stream>>>(w1, pw1, 2, 5, 16, 12, 32);
  pack_weights_kernel<<<(4 * 9  * 512 + 255) / 256, 256, 0, stream>>>(w2, pw2, 4, 9, 32, 32, 64);
  pack_weights_kernel<<<(2 * 18 * 512 + 255) / 256, 256, 0, stream>>>(w3, pw3, 2, 18, 64, 64, 24);
  bn_prep_kernel<<<1, 64, 0, stream>>>(g1, b1, m1, v1, bns1, bnb1, 32);
  bn_prep_kernel<<<1, 64, 0, stream>>>(g2, b2, m2, v2, bns2, bnb2, 64);

  // stage 1: warp + guidance pack
  guidance_kernel<<<(NPIX + 255) / 256, 256, 0, stream>>>(disp, normal, left, right, guid);

  // stage 2: WMMA convs (8 waves / block, 128-pixel row segment / block)
  conv_wmma_kernel<1, 2, 5,  16, 32, true ><<<NTILES / 8, 256, 0, stream>>>(guid, pw1, bns1, bnb1, x1);
  conv_wmma_kernel<2, 4, 9,  32, 64, true ><<<NTILES / 8, 256, 0, stream>>>(x1,   pw2, bns2, bnb2, x2);
  conv_wmma_kernel<3, 2, 18, 64, 32, false><<<NTILES / 8, 256, 0, stream>>>(x2,   pw3, nullptr, nullptr, y3);

  // stage 3: propagation
  final_kernel<<<(NPIX + 255) / 256, 256, 0, stream>>>(y3, disp, conf, affS, outp);
}
